// LSAPPNP_57604101374663
// MI455X (gfx1250) — matrix operand order repaired and checked by reading, not compiled, not run on hardware
//
#include <hip/hip_runtime.h>

#define N_NODES 100000
#define F_INF   500
#define KP      512     // F_IN padded to multiple of 64
#define HID     256
#define NCLS    40
#define NCLS_P  48      // padded to 3x16
#define ALPHA0  0.1f
#define KSTEP   64
#define ASTR    144     // LDS row stride for 64 bf16 (+16B pad), 16B aligned

typedef __bf16 bf16_t;
typedef bf16_t v16bf __attribute__((ext_vector_type(16)));
typedef float  v8f   __attribute__((ext_vector_type(8)));

// ---------------- CDNA5 async global->LDS path (guarded) ----------------
#if defined(__has_builtin)
# if __has_builtin(__builtin_amdgcn_global_load_async_to_lds_b128)
#  define HAVE_ASYNC 1
# endif
# if __has_builtin(__builtin_amdgcn_s_wait_asynccnt)
#  define HAVE_WAITA 1
# endif
#endif

#if defined(HAVE_ASYNC)
typedef int v4i __attribute__((vector_size(4 * sizeof(int))));
typedef __attribute__((address_space(1))) v4i gv4i;   // global
typedef __attribute__((address_space(3))) v4i lv4i;   // LDS
__device__ __forceinline__ void cp16(void* lds_dst, const void* gsrc) {
    // GLOBAL_LOAD_ASYNC_TO_LDS_B128: DMA 16B to LDS, tracked by ASYNCcnt
    __builtin_amdgcn_global_load_async_to_lds_b128(
        (gv4i*)gsrc, (lv4i*)lds_dst, 0, 0);
}
#else
__device__ __forceinline__ void cp16(void* lds_dst, const void* gsrc) {
    *(uint4*)lds_dst = *(const uint4*)gsrc;
}
#endif

#if defined(HAVE_WAITA)
# define WAIT_ASYNC(n) __builtin_amdgcn_s_wait_asynccnt(n)
#elif defined(HAVE_ASYNC)
# define WAIT_ASYNC_STR(n) asm volatile("s_wait_asynccnt " #n ::: "memory")
# define WAIT_ASYNC(n) WAIT_ASYNC_STR(n)
#else
# define WAIT_ASYNC(n) ((void)0)
#endif

// ---------------- workspace layout (bytes, all 256-aligned) ----------------
#define OFF_XB   0ULL                                   // bf16 [N][512]   102,400,000
#define OFF_W1T  102400000ULL                           // bf16 [256][512]     262,144
#define OFF_W2T  102662144ULL                           // bf16 [48][256]       24,576
#define OFF_HB   102686720ULL                           // bf16 [N][256]    51,200,000
#define OFF_H2   153886720ULL                           // f32  [N][40]     16,000,000
#define OFF_ACC  169886720ULL                           // f32  [N][40]     16,000,000

// ---------------- conversion / padding kernels ----------------
__global__ __launch_bounds__(256) void conv_x_bf16(const float* __restrict__ x,
                                                   bf16_t* __restrict__ xb) {
    long tid = (long)blockIdx.x * 256 + threadIdx.x;    // N*512 threads
    long row = tid >> 9;
    int  k   = (int)(tid & 511);
    float v = (k < F_INF) ? x[row * F_INF + k] : 0.0f;
    xb[tid] = (bf16_t)v;
}

__global__ __launch_bounds__(256) void conv_w1t_bf16(const float* __restrict__ W1,
                                                     bf16_t* __restrict__ w1t) {
    int tid = blockIdx.x * 256 + threadIdx.x;           // 256*512 threads
    int n = tid >> 9;
    int k = tid & 511;
    float v = (k < F_INF) ? W1[k * HID + n] : 0.0f;     // W1 is [500][256] row-major
    w1t[n * KP + k] = (bf16_t)v;
}

__global__ __launch_bounds__(256) void conv_w2t_bf16(const float* __restrict__ W2,
                                                     bf16_t* __restrict__ w2t) {
    int tid = blockIdx.x * 256 + threadIdx.x;           // 48*256 threads
    int n = tid >> 8;
    int k = tid & 255;
    float v = (n < NCLS) ? W2[k * NCLS + n] : 0.0f;     // W2 is [256][40] row-major
    w2t[n * HID + k] = (bf16_t)v;
}

// ---------------- GEMM1: h = relu(x @ W1 + b1), bf16 WMMA ----------------
// block tile 128(M) x 64(N), K staged in 64-chunks, double-buffered async LDS.
// 8 waves: 4(M) x 2(N), each wave owns a 32x32 C tile (2x2 WMMA frags).
__global__ __launch_bounds__(256) void gemm1_wmma(const bf16_t* __restrict__ xb,
                                                  const bf16_t* __restrict__ w1t,
                                                  const float*  __restrict__ b1,
                                                  bf16_t* __restrict__ hb) {
    __shared__ __align__(16) unsigned char ldsA[2][128 * ASTR];  // 128 x 64 bf16
    __shared__ __align__(16) unsigned char ldsB[2][64 * ASTR];   // 64(n) x 64(k) bf16

    const int t    = threadIdx.x;
    const int lane = t & 31;
    const int wv   = t >> 5;
    const int half = lane >> 4;
    const int l16  = lane & 15;
    const int wm   = wv >> 1;        // 0..3
    const int wn   = wv & 1;         // 0..1

    const long m0 = (long)blockIdx.x * 128;
    const int  n0 = blockIdx.y * 64;

    // stage one K-tile: A = 4 async ops/thread, B = 2 -> ASYNCcnt += 6 per wave
    auto stage = [&](int buf, int k0) {
        #pragma unroll
        for (int i = 0; i < 4; ++i) {
            int c = t + i * 256;
            int row = c >> 3, seg = c & 7;
            long gr = m0 + row;
            if (gr > N_NODES - 1) gr = N_NODES - 1;   // clamp: rows >= N never stored
            cp16(ldsA[buf] + row * ASTR + seg * 16,
                 xb + gr * KP + k0 + seg * 8);
        }
        #pragma unroll
        for (int i = 0; i < 2; ++i) {
            int c = t + i * 256;
            int n = c >> 3, seg = c & 7;
            cp16(ldsB[buf] + n * ASTR + seg * 16,
                 w1t + (long)(n0 + n) * KP + k0 + seg * 8);
        }
    };

    v8f acc[2][2] = {};

    stage(0, 0);
    for (int it = 0; it < KP / KSTEP; ++it) {
        const int cur = it & 1;
        if (it + 1 < KP / KSTEP) {
            stage(cur ^ 1, (it + 1) * KSTEP);   // prefetch next tile
            WAIT_ASYNC(6);                      // in-order: tile `it` resident
        } else {
            WAIT_ASYNC(0);
        }
        __syncthreads();                        // tile visible across all waves

        #pragma unroll
        for (int kk = 0; kk < KSTEP; kk += 32) {
            v16bf af[2], bfr[2];
            #pragma unroll
            for (int mi = 0; mi < 2; ++mi) {
                const unsigned char* p = ldsA[cur] + (wm * 32 + mi * 16 + l16) * ASTR
                                       + kk * 2 + half * 16;
                ((uint4*)&af[mi])[0] = *(const uint4*)(p);
                ((uint4*)&af[mi])[1] = *(const uint4*)(p + 32);
            }
            #pragma unroll
            for (int ni = 0; ni < 2; ++ni) {
                const unsigned char* p = ldsB[cur] + (wn * 32 + ni * 16 + l16) * ASTR
                                       + kk * 2 + half * 16;
                ((uint4*)&bfr[ni])[0] = *(const uint4*)(p);
                ((uint4*)&bfr[ni])[1] = *(const uint4*)(p + 32);
            }
            #pragma unroll
            for (int mi = 0; mi < 2; ++mi)
                #pragma unroll
                for (int ni = 0; ni < 2; ++ni)
                    acc[mi][ni] = __builtin_amdgcn_wmma_f32_16x16x32_bf16(
                        false, af[mi], false, bfr[ni],
                        (short)0, acc[mi][ni], false, false);
        }
        __syncthreads();        // all waves done reading `cur` before it is rewritten
    }

    // epilogue: bias + relu + bf16 store.  C layout: lane%16 = N, M = half*8 + r
    #pragma unroll
    for (int mi = 0; mi < 2; ++mi)
        #pragma unroll
        for (int ni = 0; ni < 2; ++ni) {
            int col = n0 + wn * 32 + ni * 16 + l16;
            float bias = b1[col];
            #pragma unroll
            for (int r = 0; r < 8; ++r) {
                long row = m0 + wm * 32 + mi * 16 + half * 8 + r;
                if (row < N_NODES) {
                    float v = acc[mi][ni][r] + bias;
                    v = v > 0.0f ? v : 0.0f;
                    hb[row * HID + col] = (bf16_t)v;
                }
            }
        }
}

// ---------------- GEMM2: h2 = h @ W2 + b2 ; acc = 0.1*h2 ----------------
// block tile 128(M) x 48(N); w2t fully LDS-resident; A double-buffered async.
__global__ __launch_bounds__(256) void gemm2_wmma(const bf16_t* __restrict__ hb,
                                                  const bf16_t* __restrict__ w2t,
                                                  const float*  __restrict__ b2,
                                                  float* __restrict__ h2,
                                                  float* __restrict__ accb) {
    __shared__ __align__(16) unsigned char ldsA[2][128 * ASTR];  // 128 x 64 bf16
    __shared__ __align__(16) unsigned char ldsB[48 * 528];       // 48 x 256 bf16

    const int t    = threadIdx.x;
    const int lane = t & 31;
    const int wv   = t >> 5;
    const int half = lane >> 4;
    const int l16  = lane & 15;
    const long m0  = (long)blockIdx.x * 128;

    // preload entire w2t: 1536 x 16B chunks -> 6 async ops per wave
    #pragma unroll
    for (int i = 0; i < 6; ++i) {
        int c = t + i * 256;
        int n = c >> 5, seg = c & 31;
        cp16(ldsB + n * 528 + seg * 16, w2t + n * HID + seg * 8);
    }

    auto stageA = [&](int buf, int k0) {   // 4 async ops per wave
        #pragma unroll
        for (int i = 0; i < 4; ++i) {
            int c = t + i * 256;
            int row = c >> 3, seg = c & 7;
            long gr = m0 + row;
            if (gr > N_NODES - 1) gr = N_NODES - 1;
            cp16(ldsA[buf] + row * ASTR + seg * 16,
                 hb + gr * HID + k0 + seg * 8);
        }
    };

    v8f acc[3] = {};

    stageA(0, 0);
    for (int it = 0; it < HID / KSTEP; ++it) {
        const int cur = it & 1;
        if (it + 1 < HID / KSTEP) {
            stageA(cur ^ 1, (it + 1) * KSTEP);
            WAIT_ASYNC(4);                 // in-order: B preload + tile `it` resident
        } else {
            WAIT_ASYNC(0);
        }
        __syncthreads();

        #pragma unroll
        for (int kk = 0; kk < KSTEP; kk += 32) {
            const int kglob = it * KSTEP + kk;
            v16bf af;
            const unsigned char* pa = ldsA[cur] + (wv * 16 + l16) * ASTR
                                    + kk * 2 + half * 16;
            ((uint4*)&af)[0] = *(const uint4*)(pa);
            ((uint4*)&af)[1] = *(const uint4*)(pa + 32);

            #pragma unroll
            for (int ni = 0; ni < 3; ++ni) {
                v16bf bfr;
                const unsigned char* pb = ldsB + (ni * 16 + l16) * 528
                                        + (kglob + half * 8) * 2;
                ((uint4*)&bfr)[0] = *(const uint4*)(pb);
                ((uint4*)&bfr)[1] = *(const uint4*)(pb + 32);
                acc[ni] = __builtin_amdgcn_wmma_f32_16x16x32_bf16(
                    false, af, false, bfr, (short)0, acc[ni], false, false);
            }
        }
        __syncthreads();
    }

    #pragma unroll
    for (int ni = 0; ni < 3; ++ni) {
        int col = ni * 16 + l16;
        if (col < NCLS) {
            float bias = b2[col];
            #pragma unroll
            for (int r = 0; r < 8; ++r) {
                long row = m0 + wv * 16 + half * 8 + r;
                if (row < N_NODES) {
                    float v = acc[ni][r] + bias;
                    h2[row * NCLS + col]   = v;
                    accb[row * NCLS + col] = ALPHA0 * v;
                }
            }
        }
    }
}

// ---------------- edge scatter: acc[r] += val * h2[c] ----------------
// 8 threads per edge, 5 classes each (40 total). h2 (16MB) stays L2-resident.
__global__ __launch_bounds__(256) void scatter_spmm(const int*   __restrict__ erow,
                                                    const int*   __restrict__ ecol,
                                                    const float* __restrict__ eval,
                                                    const float* __restrict__ h2,
                                                    float* __restrict__ accb, int E) {
    long tid = (long)blockIdx.x * 256 + threadIdx.x;
    long e   = tid >> 3;
    if (e >= E) return;
    int  part = (int)(tid & 7);
    int  r = erow[e], c = ecol[e];
    float v = eval[e];
    int  cb = part * 5;
    const float* src = h2   + (long)c * NCLS + cb;
    float*       dst = accb + (long)r * NCLS + cb;
    #pragma unroll
    for (int i = 0; i < 5; ++i)
        atomicAdd(dst + i, v * src[i]);
}

// ---------------- row-wise log_softmax over 40 classes ----------------
__global__ __launch_bounds__(256) void row_logsoftmax(const float* __restrict__ accb,
                                                      float* __restrict__ out) {
    long row = (long)blockIdx.x * 256 + threadIdx.x;
    if (row >= N_NODES) return;
    const float4* p = (const float4*)(accb + row * NCLS);
    float4 v[10];
    float m = -3.4e38f;
    #pragma unroll
    for (int i = 0; i < 10; ++i) {
        v[i] = p[i];
        m = fmaxf(m, fmaxf(fmaxf(v[i].x, v[i].y), fmaxf(v[i].z, v[i].w)));
    }
    float s = 0.0f;
    #pragma unroll
    for (int i = 0; i < 10; ++i)
        s += __expf(v[i].x - m) + __expf(v[i].y - m) +
             __expf(v[i].z - m) + __expf(v[i].w - m);
    float lse = m + __logf(s);
    float4* o = (float4*)(out + row * NCLS);
    #pragma unroll
    for (int i = 0; i < 10; ++i) {
        float4 w = v[i];
        w.x -= lse; w.y -= lse; w.z -= lse; w.w -= lse;
        o[i] = w;
    }
}

// ---------------- launcher ----------------
extern "C" void kernel_launch(void* const* d_in, const int* in_sizes, int n_in,
                              void* d_out, int out_size, void* d_ws, size_t ws_size,
                              hipStream_t stream) {
    const float* x    = (const float*)d_in[0];
    const float* W1   = (const float*)d_in[1];
    const float* b1   = (const float*)d_in[2];
    const float* W2   = (const float*)d_in[3];
    const float* b2   = (const float*)d_in[4];
    const int*   erow = (const int*)d_in[5];
    const int*   ecol = (const int*)d_in[6];
    const float* eval = (const float*)d_in[7];
    const int    E    = in_sizes[5];

    char* ws = (char*)d_ws;
    bf16_t* xb   = (bf16_t*)(ws + OFF_XB);
    bf16_t* w1t  = (bf16_t*)(ws + OFF_W1T);
    bf16_t* w2t  = (bf16_t*)(ws + OFF_W2T);
    bf16_t* hb   = (bf16_t*)(ws + OFF_HB);
    float*  h2   = (float*) (ws + OFF_H2);
    float*  accb = (float*) (ws + OFF_ACC);

    conv_x_bf16  <<<(N_NODES * 512) / 256, 256, 0, stream>>>(x, xb);
    conv_w1t_bf16<<<(HID * KP) / 256,      256, 0, stream>>>(W1, w1t);
    conv_w2t_bf16<<<(NCLS_P * HID) / 256,  256, 0, stream>>>(W2, w2t);

    dim3 g1((N_NODES + 127) / 128, HID / 64);
    gemm1_wmma<<<g1, 256, 0, stream>>>(xb, w1t, b1, hb);

    gemm2_wmma<<<(N_NODES + 127) / 128, 256, 0, stream>>>(hb, w2t, b2, h2, accb);

    long nthr = (long)E * 8;
    scatter_spmm<<<(unsigned)((nthr + 255) / 256), 256, 0, stream>>>(erow, ecol, eval,
                                                                    h2, accb, E);

    row_logsoftmax<<<(N_NODES + 255) / 256, 256, 0, stream>>>(accb, (float*)d_out);
}
